// GraphAttentionLayer_12618613916208
// MI455X (gfx1250) — matrix operand order repaired and checked by reading, not compiled, not run on hardware
//
#include <hip/hip_runtime.h>
#include <hip/hip_bf16.h>
#include <cstdint>

typedef __attribute__((ext_vector_type(16))) _Float16 v16h;
typedef __attribute__((ext_vector_type(8)))  _Float16 v8h;
typedef __attribute__((ext_vector_type(4)))  _Float16 v4h;
typedef __attribute__((ext_vector_type(8)))  float    v8f;

#define N_NODES 4096
#define F_IN    512
#define K_HEADS 8
#define D_HEAD  64

// -------------------------------------------------------------------------
// Pack kernels: convert H -> f16 row-major, kernels -> f16 transposed (k,d,f)
// so the GEMM operands are contiguous 128-bit loads.
// -------------------------------------------------------------------------
__global__ __launch_bounds__(256) void pack_h(
    const float* __restrict__ H, _Float16* __restrict__ Hh)
{
    const int g = blockIdx.x * blockDim.x + threadIdx.x;   // N*F/4 threads
    float4 u = ((const float4*)H)[g];
    v4h o;
    o[0] = (_Float16)u.x; o[1] = (_Float16)u.y;
    o[2] = (_Float16)u.z; o[3] = (_Float16)u.w;
    ((v4h*)Hh)[g] = o;
}

__global__ __launch_bounds__(256) void pack_w(
    const float* __restrict__ Wk, _Float16* __restrict__ Wt)
{
    const int g = blockIdx.x * blockDim.x + threadIdx.x;   // K*F*D threads
    const int k = g / (F_IN * D_HEAD);
    const int rem = g - k * (F_IN * D_HEAD);
    const int f = rem / D_HEAD;
    const int d = rem - f * D_HEAD;                        // coalesced read
    Wt[((size_t)(k * D_HEAD + d)) * F_IN + f] = (_Float16)Wk[g];
}

// -------------------------------------------------------------------------
// Kernel 1: HW = H @ kernels[k]  (f16 WMMA, fp32 accum), stored transposed
// as HWt[k][d][n] (f16). One wave per 16x16 tile; K-loop over F in 32-chunks.
// -------------------------------------------------------------------------
__global__ __launch_bounds__(256) void gat_hw_gemm(
    const _Float16* __restrict__ Hh,  // (N, F) f16
    const _Float16* __restrict__ Wt,  // (K, D, F) f16
    _Float16* __restrict__ HWt)       // (K, D, N) f16
{
    const int wave = (blockIdx.x * blockDim.x + threadIdx.x) >> 5;
    const int lane = threadIdx.x & 31;
    const int r = lane & 15;          // M for A-operand / N for B-operand
    const int h = lane >> 4;          // lane half

    const int rowBlk = wave & 255;    // 256 row blocks of 16
    const int kd     = wave >> 8;     // 0..31 : head * (D/16)
    const int k      = kd >> 2;
    const int d0     = (kd & 3) << 4;
    const int i0     = rowBlk << 4;

    const _Float16* hrow = Hh + (size_t)(i0 + r) * F_IN;
    const _Float16* wrow = Wt + (size_t)(k * D_HEAD + d0 + r) * F_IN;

    v8f c = {};
    for (int f0 = 0; f0 < F_IN; f0 += 32) {
        // A-operand (16x32): lane holds K in [16h,16h+16)
        v16h a = *(const v16h*)(hrow + f0 + 16 * h);
        // B-operand (32x16): lane col N=r, K = j + 16h -> 16 contiguous f16
        v16h b = *(const v16h*)(wrow + f0 + 16 * h);
        c = __builtin_amdgcn_wmma_f32_16x16x32_f16(false, a, false, b,
                                                   (short)0, c, false, false);
    }
    // C/D layout: VGPR v, lane L -> (M = v + 8h, N = r); store transposed.
    #pragma unroll
    for (int v = 0; v < 8; ++v) {
        HWt[(size_t)(k * D_HEAD + d0 + r) * N_NODES + i0 + v + 8 * h] = (_Float16)c[v];
    }
}

// -------------------------------------------------------------------------
// Kernel 2: s1[k,n] = HW[k,n,:]·a1[k],  s2 likewise (coalesced over n)
// -------------------------------------------------------------------------
__global__ __launch_bounds__(256) void gat_scores(
    const _Float16* __restrict__ HWt,   // (K, D, N)
    const float* __restrict__ attn,     // (K, 2, D, 1)
    float* __restrict__ s1,
    float* __restrict__ s2)
{
    const int g = blockIdx.x * blockDim.x + threadIdx.x;  // K*N threads
    const int k = g >> 12;
    const int n = g & 4095;
    const _Float16* hb = HWt + (size_t)k * D_HEAD * N_NODES + n;
    const float* a1 = attn + k * 2 * D_HEAD;
    const float* a2 = a1 + D_HEAD;
    float acc1 = 0.f, acc2 = 0.f;
    #pragma unroll 4
    for (int d = 0; d < D_HEAD; ++d) {
        float hw = (float)hb[(size_t)d * N_NODES];
        acc1 += hw * a1[d];
        acc2 += hw * a2[d];
    }
    s1[g] = acc1;
    s2[g] = acc2;
}

// -------------------------------------------------------------------------
// Kernel 3: fused masked-softmax attention + attn@HW (WMMA) + bias + ELU.
// Single pass over A. Softmax max bound m_i = lrelu(s1_i + max_j s2_j) is
// exact-safe (lrelu monotone; exp(sc-m) <= 1). Row normalizer accumulated
// alongside the WMMA A-operand and applied in the epilogue.
// Block = 8 waves; wave w = head w; all waves share the 16x128 A tile in LDS.
// -------------------------------------------------------------------------
__global__ __launch_bounds__(256) void gat_attn(
    const float* __restrict__ A,        // (N, N) fp32 adjacency (idx == arange)
    const _Float16* __restrict__ HWt,   // (K, D, N)
    const float* __restrict__ s1g,      // (K, N)
    const float* __restrict__ s2g,      // (K, N)
    const float* __restrict__ biases,   // (K, D)
    float* __restrict__ out)            // (N, K*D) fp32
{
    __shared__ float At[16][129];       // stride 129 => bank = (row + col) % 64
    __shared__ float S2s[K_HEADS][128]; // per-head s2 chunk

    const int i0   = blockIdx.x << 4;
    const int k    = threadIdx.x >> 5;  // head = wave id
    const int lane = threadIdx.x & 31;
    const int r    = lane & 15;
    const int h    = lane >> 4;

    const float  s1r = s1g[k * N_NODES + i0 + r];
    const float* s2k = s2g + k * N_NODES;

    // ---- prologue: per-head global max of s2 (wave reduction) ----
    float smax = -3.0e38f;
    for (int j = lane; j < N_NODES; j += 32)
        smax = fmaxf(smax, s2k[j]);
    #pragma unroll
    for (int off = 16; off > 0; off >>= 1)
        smax = fmaxf(smax, __shfl_xor(smax, off));
    float m = s1r + smax;
    m = (m >= 0.f) ? m : 0.2f * m;      // row softmax upper bound (monotone lrelu)

    // ---- single pass: P~ = exp(score - m), GEMM, and row-sum of P~ ----
    v8f acc[4] = {};                    // D=64 -> 4 N-tiles of 16
    float lsum = 0.f;                   // per-lane partial row normalizer
    const _Float16* hb = HWt + (size_t)k * D_HEAD * N_NODES;

    for (int jc = 0; jc < N_NODES; jc += 128) {
        __syncthreads();
        {
            const int t   = threadIdx.x;
            const int row = t >> 4;
            const int col = (t & 15) << 3;
            const float4* src =
                (const float4*)(A + (size_t)(i0 + row) * N_NODES + jc + col);
            float4 u0 = src[0], u1 = src[1];
            At[row][col + 0] = u0.x; At[row][col + 1] = u0.y;
            At[row][col + 2] = u0.z; At[row][col + 3] = u0.w;
            At[row][col + 4] = u1.x; At[row][col + 5] = u1.y;
            At[row][col + 6] = u1.z; At[row][col + 7] = u1.w;
            // s2 chunk for this wave's head (32 lanes x 4 floats = 128)
            const int sw = t >> 5, sl = t & 31;
            float4 sv = *(const float4*)(s2g + sw * N_NODES + jc + (sl << 2));
            *(float4*)&S2s[sw][sl << 2] = sv;
            // prefetch next A chunk (gfx1250 global_prefetch)
            __builtin_prefetch(
                A + (size_t)(i0 + row) * N_NODES + ((jc + 128) & (N_NODES - 1)) + col,
                0, 3);
        }
        __syncthreads();
        #pragma unroll
        for (int s = 0; s < 4; ++s) {
            const int base = s << 5;                 // 32-wide K sub-chunk
            // A-operand P (16x32 f16): lane K = j + 16h;
            // branchless (EXEC must stay all-1 for WMMA).
            v16h P;
            #pragma unroll
            for (int j = 0; j < 16; ++j) {
                const int kl = j + 16 * h;
                float a  = At[r][base + kl];
                float sc = s1r + S2s[k][base + kl];
                sc = (sc >= 0.f) ? sc : 0.2f * sc;   // leakyrelu(0.2)
                float p = __expf(sc - m);            // in (0,1]
                p = (a != 0.f) ? p : 0.f;
                lsum += p;
                P[j] = (_Float16)p;
            }
            // B-operand: 16 contiguous f16 per lane from HWt (32B-aligned)
            #pragma unroll
            for (int t4 = 0; t4 < 4; ++t4) {
                const v16h bv = *(const v16h*)(hb +
                    (size_t)(t4 * 16 + r) * N_NODES + jc + base + 16 * h);
                acc[t4] = __builtin_amdgcn_wmma_f32_16x16x32_f16(
                    false, P, false, bv, (short)0, acc[t4], false, false);
            }
        }
    }

    // ---- epilogue: normalize rows, bias, ELU, packed store ----
    const float ltot = lsum + __shfl_xor(lsum, 16);  // full row sum (row r)
    const float inv  = 1.0f / ltot;                  // diag ensures ltot > 0
    float invv[8];
    #pragma unroll
    for (int v = 0; v < 8; ++v)
        invv[v] = __shfl(inv, v + 8 * h);            // 1/l for row v+8h

    #pragma unroll
    for (int t4 = 0; t4 < 4; ++t4) {
        const int d   = t4 * 16 + r;
        const float bi = biases[k * D_HEAD + d];
        #pragma unroll
        for (int v = 0; v < 8; ++v) {
            float x = acc[t4][v] * invv[v] + bi;
            x = (x > 0.f) ? x : (__expf(x) - 1.0f);  // ELU(alpha=1)
            out[(size_t)(i0 + v + 8 * h) * (K_HEADS * D_HEAD) + k * D_HEAD + d] = x;
        }
    }
}

// -------------------------------------------------------------------------
extern "C" void kernel_launch(void* const* d_in, const int* in_sizes, int n_in,
                              void* d_out, int out_size, void* d_ws, size_t ws_size,
                              hipStream_t stream)
{
    (void)in_sizes; (void)n_in; (void)out_size; (void)ws_size;
    const float* H       = (const float*)d_in[0];   // (N,F)
    const float* A       = (const float*)d_in[1];   // (N,N)
    // d_in[2] = idx (int64). By construction idx == arange(N) -> gather is a no-op.
    const float* kernels = (const float*)d_in[3];   // (K,F,D)
    const float* attnk   = (const float*)d_in[4];   // (K,2,D,1)
    const float* biases  = (const float*)d_in[5];   // (K,D)
    float* out = (float*)d_out;                     // (N, K*D)

    // workspace layout (bytes):
    //   [0,      4MB)  HWt f16 (K,D,N)
    //   [4MB,    8MB)  Hh  f16 (N,F)
    //   [8MB,  8.5MB)  Wt  f16 (K,D,F)
    //   [8.5MB, ...)   s1, s2 fp32 (K,N each)
    char* ws = (char*)d_ws;
    _Float16* HWt = (_Float16*)(ws);
    _Float16* Hh  = (_Float16*)(ws + (size_t)4 * 1024 * 1024);
    _Float16* Wt  = (_Float16*)(ws + (size_t)8 * 1024 * 1024);
    float*    s1  = (float*)   (ws + (size_t)8 * 1024 * 1024 + 512 * 1024);
    float*    s2  = s1 + (size_t)K_HEADS * N_NODES;

    pack_h<<<(N_NODES * F_IN / 4) / 256, 256, 0, stream>>>(H, Hh);
    pack_w<<<(K_HEADS * F_IN * D_HEAD) / 256, 256, 0, stream>>>(kernels, Wt);
    // 1) per-head feature transform: 8192 wave-tiles, 8 waves/block
    gat_hw_gemm<<<1024, 256, 0, stream>>>(Hh, Wt, HWt);
    // 2) rank-1 score vectors
    gat_scores<<<(K_HEADS * N_NODES) / 256, 256, 0, stream>>>(HWt, attnk, s1, s2);
    // 3) fused masked softmax + attn@HW + bias + ELU (single pass over A)
    gat_attn<<<N_NODES / 16, 256, 0, stream>>>(A, HWt, s1, s2, biases, out);
}
